// CBContrastiveLoss_72095321030692
// MI455X (gfx1250) — matrix operand order repaired and checked by reading, not compiled, not run on hardware
//
#include <hip/hip_runtime.h>
#include <math.h>

// ---------------------------------------------------------------------------
// CBContrastiveLoss on gfx1250 (MI455X), wave32 + WMMA f16 (f32 accumulate).
//   S = Fn * Fn^T / tau   (Fn = row-normalized features, kept as f16 in ws)
//   pass1: rowSum[i] = sum_j!=i exp(s_ij - M),  M = 1/tau (fixed max bound,
//          valid since |dot| <= 1 for normalized rows)  -> logZ
//   pass2: acc[i]    = sum_{pos j} lp*(1-exp(lp))^2,  lp = s_ij - logZ_i
//   final: out = -(1/N) * sum_i [cnt_i>0] * acc[i]/cnt_i * cw[lab_i]
// ---------------------------------------------------------------------------

typedef _Float16 v16h __attribute__((ext_vector_type(16)));
typedef _Float16 v8h  __attribute__((ext_vector_type(8)));
typedef _Float16 v4h  __attribute__((ext_vector_type(4)));
typedef float    v8f  __attribute__((ext_vector_type(8)));

#define N_ROWS 8192
#define DIMS   512
#define KCH    (DIMS / 32)        // 16 K-chunks of 32 per 16x16 tile
#define INV_T  14.285714285714286f
#define MAXB   INV_T
#define JSPLIT 8                  // waves per strip (column split)
#define JCHUNK (N_ROWS / 16 / JSPLIT) // 64 column tiles per wave

union Frag { v16h v; v8h h[2]; };

__device__ __forceinline__ float groupReduce16(float x) {
  x += __shfl_xor(x, 1, 16);
  x += __shfl_xor(x, 2, 16);
  x += __shfl_xor(x, 4, 16);
  x += __shfl_xor(x, 8, 16);
  return x;
}

__device__ __forceinline__ void loadB(Frag& b, const _Float16* __restrict__ fh,
                                      int j0, int mn, int half, int kc) {
  const _Float16* brow = fh + (size_t)(j0 + mn) * DIMS + half * 8 + kc * 32;
  b.h[0] = *(const v8h*)(brow);
  b.h[1] = *(const v8h*)(brow + 16);
}

__device__ __forceinline__ void prefetchTile(const _Float16* __restrict__ fh,
                                             int j0, int mn, int half) {
  // next tile B panel: 16 rows x 1024B; 32 lanes x 512B each, 128B strides
  const char* p = (const char*)(fh + (size_t)(j0 + mn) * DIMS) + half * 512;
  __builtin_prefetch(p, 0, 3);
  __builtin_prefetch(p + 128, 0, 3);
  __builtin_prefetch(p + 256, 0, 3);
  __builtin_prefetch(p + 384, 0, 3);
}

// --- row-normalize f32 -> f16, one wave per 512-elem row ---------------------
__global__ void k_normalize(const float* __restrict__ f, _Float16* __restrict__ fh) {
  const int wave = threadIdx.x >> 5;
  const int lane = threadIdx.x & 31;
  const int row  = blockIdx.x * (blockDim.x >> 5) + wave;
  if (row >= N_ROWS) return;
  const float4* src = (const float4*)(f + (size_t)row * DIMS);
  float4 x[4];
  float ss = 0.f;
#pragma unroll
  for (int t = 0; t < 4; ++t) {
    x[t] = src[lane + 32 * t];
    ss += x[t].x * x[t].x + x[t].y * x[t].y + x[t].z * x[t].z + x[t].w * x[t].w;
  }
#pragma unroll
  for (int off = 16; off; off >>= 1) ss += __shfl_xor(ss, off, 32);
  const float inv = 1.0f / sqrtf(ss);
  v4h* dst = (v4h*)(fh + (size_t)row * DIMS);
#pragma unroll
  for (int t = 0; t < 4; ++t) {
    v4h o;
    o[0] = (_Float16)(x[t].x * inv);
    o[1] = (_Float16)(x[t].y * inv);
    o[2] = (_Float16)(x[t].z * inv);
    o[3] = (_Float16)(x[t].w * inv);
    dst[lane + 32 * t] = o;
  }
}

// --- label histogram (9 classes) --------------------------------------------
__global__ void k_hist(const int* __restrict__ labels, int* __restrict__ hist) {
  const int i = blockIdx.x * blockDim.x + threadIdx.x;
  if (i < N_ROWS) atomicAdd(&hist[labels[i]], 1);
}

// --- pass 1: row sum of exp(s - M), diagonal excluded ------------------------
__global__ __launch_bounds__(256) void k_pass1(const _Float16* __restrict__ fh,
                                               float* __restrict__ rowSum) {
  const int wave = threadIdx.x >> 5;
  const int lane = threadIdx.x & 31;
  const int half = lane >> 4;     // lane group (0: lanes 0-15, 1: lanes 16-31)
  const int mn   = lane & 15;     // A row within tile / B column within tile
  const int i0   = blockIdx.x * 16;

  // Preload the wave's full A strip: 16 rows x 512 K  (128 VGPRs)
  const _Float16* arow = fh + (size_t)(i0 + mn) * DIMS;
  Frag A[KCH];
#pragma unroll
  for (int kc = 0; kc < KCH; ++kc) {
    A[kc].h[0] = *(const v8h*)(arow + kc * 32 + half * 8);
    A[kc].h[1] = *(const v8h*)(arow + kc * 32 + 16 + half * 8);
  }

  float rs[8] = {0.f, 0.f, 0.f, 0.f, 0.f, 0.f, 0.f, 0.f};
  const int jt0 = wave * JCHUNK, jt1 = jt0 + JCHUNK;

  Frag b[2];
  loadB(b[0], fh, jt0 * 16, mn, half, 0);   // prologue: chunk 0 of first tile
  for (int jt = jt0; jt < jt1; ++jt) {
    const int j0  = jt * 16;
    const int nj0 = (jt + 1 < jt1) ? j0 + 16 : j0;  // clamped next tile
    prefetchTile(fh, nj0, mn, half);
    v8f c0 = {}, c1 = {};
#pragma unroll
    for (int kc = 0; kc < KCH; ++kc) {
      // prefetch chunk kc+1 (or next tile's chunk 0) into the other buffer
      if (kc + 1 < KCH) loadB(b[(kc + 1) & 1], fh, j0, mn, half, kc + 1);
      else              loadB(b[(kc + 1) & 1], fh, nj0, mn, half, 0);
      if (kc & 1)
        c1 = __builtin_amdgcn_wmma_f32_16x16x32_f16(false, A[kc].v, false, b[kc & 1].v,
                                                    (short)0, c1, false, false);
      else
        c0 = __builtin_amdgcn_wmma_f32_16x16x32_f16(false, A[kc].v, false, b[kc & 1].v,
                                                    (short)0, c0, false, false);
    }
    const v8f c = c0 + c1;
    const bool diag = (j0 == i0);
#pragma unroll
    for (int v = 0; v < 8; ++v) {
      float ex = __expf(c[v] * INV_T - MAXB);
      if (diag && (v + 8 * half) == mn) ex = 0.f; // mask self-pair
      rs[v] += ex;
    }
  }
#pragma unroll
  for (int v = 0; v < 8; ++v) {
    const float r = groupReduce16(rs[v]);
    if (mn == 0) atomicAdd(&rowSum[i0 + v + 8 * half], r);
  }
}

// --- logZ[i] = log(rowSum[i]) + M -------------------------------------------
__global__ void k_logz(const float* __restrict__ rowSum, float* __restrict__ logZ) {
  const int i = blockIdx.x * blockDim.x + threadIdx.x;
  if (i < N_ROWS) logZ[i] = logf(rowSum[i]) + MAXB;
}

// --- pass 2: focal-weighted log-prob over positive pairs ---------------------
__global__ __launch_bounds__(256) void k_pass2(const _Float16* __restrict__ fh,
                                               const int* __restrict__ labels,
                                               const float* __restrict__ logZ,
                                               float* __restrict__ rowAcc) {
  const int wave = threadIdx.x >> 5;
  const int lane = threadIdx.x & 31;
  const int half = lane >> 4;
  const int mn   = lane & 15;
  const int i0   = blockIdx.x * 16;

  const _Float16* arow = fh + (size_t)(i0 + mn) * DIMS;
  Frag A[KCH];
#pragma unroll
  for (int kc = 0; kc < KCH; ++kc) {
    A[kc].h[0] = *(const v8h*)(arow + kc * 32 + half * 8);
    A[kc].h[1] = *(const v8h*)(arow + kc * 32 + 16 + half * 8);
  }

  int   labI[8];
  float lz[8];
#pragma unroll
  for (int v = 0; v < 8; ++v) {
    const int row = i0 + v + 8 * half;
    labI[v] = labels[row];
    lz[v]   = logZ[row];
  }

  float acc[8] = {0.f, 0.f, 0.f, 0.f, 0.f, 0.f, 0.f, 0.f};
  const int jt0 = wave * JCHUNK, jt1 = jt0 + JCHUNK;

  Frag b[2];
  loadB(b[0], fh, jt0 * 16, mn, half, 0);
  for (int jt = jt0; jt < jt1; ++jt) {
    const int j0  = jt * 16;
    const int nj0 = (jt + 1 < jt1) ? j0 + 16 : j0;
    prefetchTile(fh, nj0, mn, half);
    const int labj = labels[j0 + mn];  // this lane's column label
    v8f c0 = {}, c1 = {};
#pragma unroll
    for (int kc = 0; kc < KCH; ++kc) {
      if (kc + 1 < KCH) loadB(b[(kc + 1) & 1], fh, j0, mn, half, kc + 1);
      else              loadB(b[(kc + 1) & 1], fh, nj0, mn, half, 0);
      if (kc & 1)
        c1 = __builtin_amdgcn_wmma_f32_16x16x32_f16(false, A[kc].v, false, b[kc & 1].v,
                                                    (short)0, c1, false, false);
      else
        c0 = __builtin_amdgcn_wmma_f32_16x16x32_f16(false, A[kc].v, false, b[kc & 1].v,
                                                    (short)0, c0, false, false);
    }
    const v8f c = c0 + c1;
    const bool diag = (j0 == i0);
#pragma unroll
    for (int v = 0; v < 8; ++v) {
      const bool pos = (labj == labI[v]) && !(diag && (v + 8 * half) == mn);
      if (pos) {
        const float lp = c[v] * INV_T - lz[v];
        const float p  = __expf(lp);
        const float om = 1.f - p;
        acc[v] += lp * om * om;
      }
    }
  }
#pragma unroll
  for (int v = 0; v < 8; ++v) {
    const float r = groupReduce16(acc[v]);
    if (mn == 0) atomicAdd(&rowAcc[i0 + v + 8 * half], r);
  }
}

// --- final scalar reduction --------------------------------------------------
__global__ void k_final(const int* __restrict__ labels, const float* __restrict__ cw,
                        const int* __restrict__ hist, const float* __restrict__ rowAcc,
                        float* __restrict__ out) {
  const int i = blockIdx.x * blockDim.x + threadIdx.x;
  float val = 0.f;
  if (i < N_ROWS) {
    const int lab = labels[i];
    const int cnt = hist[lab] - 1;
    if (cnt > 0) val = rowAcc[i] / (float)cnt * cw[lab];
  }
#pragma unroll
  for (int off = 16; off; off >>= 1) val += __shfl_xor(val, off, 32);
  if ((threadIdx.x & 31) == 0) atomicAdd(out, -val / (float)N_ROWS);
}

// ---------------------------------------------------------------------------
extern "C" void kernel_launch(void* const* d_in, const int* in_sizes, int n_in,
                              void* d_out, int out_size, void* d_ws, size_t ws_size,
                              hipStream_t stream) {
  const float* features = (const float*)d_in[0];   // [8192, 512] f32
  const int*   labels   = (const int*)d_in[1];     // [8192]
  const float* cw       = (const float*)d_in[2];   // [9]
  float*       out      = (float*)d_out;           // scalar

  // workspace layout
  char* ws = (char*)d_ws;
  _Float16* fh = (_Float16*)ws;                                   // 8 MB
  size_t off = (size_t)N_ROWS * DIMS * sizeof(_Float16);
  float* rowSum = (float*)(ws + off); off += (size_t)N_ROWS * 4;  // zeroed
  float* rowAcc = (float*)(ws + off); off += (size_t)N_ROWS * 4;  // zeroed
  int*   hist   = (int*)(ws + off);   off += 64;                  // zeroed
  float* logZ   = (float*)(ws + off); off += (size_t)N_ROWS * 4;

  hipMemsetAsync(out, 0, sizeof(float), stream);
  hipMemsetAsync(rowSum, 0, (size_t)N_ROWS * 4 * 2 + 64, stream); // rowSum+rowAcc+hist

  k_normalize<<<N_ROWS / 8, 256, 0, stream>>>(features, fh);
  k_hist<<<N_ROWS / 256, 256, 0, stream>>>(labels, hist);
  k_pass1<<<N_ROWS / 16, 256, 0, stream>>>(fh, rowSum);
  k_logz<<<N_ROWS / 256, 256, 0, stream>>>(rowSum, logZ);
  k_pass2<<<N_ROWS / 16, 256, 0, stream>>>(fh, labels, logZ, rowAcc);
  k_final<<<N_ROWS / 256, 256, 0, stream>>>(labels, cw, hist, rowAcc, out);
}
